// QuantumHybridResNet_51745765982252
// MI455X (gfx1250) — matrix operand order repaired and checked by reading, not compiled, not run on hardware
//
#include <hip/hip_runtime.h>
#include <hip/hip_bf16.h>
#include <math.h>

// ---------------------------------------------------------------------------
// CDNA5 (gfx1250) types for WMMA
// ---------------------------------------------------------------------------
typedef _Float16 v16h __attribute__((ext_vector_type(16)));
typedef float    v8f  __attribute__((ext_vector_type(8)));
typedef float    f32x4 __attribute__((ext_vector_type(4)));

union Frag16 {             // 16 halves = 32 bytes, loaded as two 16B chunks
    v16h  v;
    f32x4 q[2];
};

#define TK 32
#define LDA 40             // padded LDS K-stride (halves): 80B -> conflict-free b128

// ---------------------------------------------------------------------------
// Implicit-GEMM conv + fused BN (+residual) (+ReLU), f16 WMMA, f32 accumulate.
//   GEMM view: A[M,K] = im2col(X), B[K,N] = W (OIHW is [N][K], K-contiguous)
//   BM x TNt tile, BM/16 waves (wave w: rows [16w,16w+16), all TNt cols).
//   LDS double-buffered; next chunk's global gather overlaps current WMMAs.
//   Per-thread A mapping fixed: row = tid % BM, k-span = (tid/BM)*16 + [0,16)
//   -> M-decode hoisted out of the K loop, K-decode stepped incrementally.
// ---------------------------------------------------------------------------
template <int BM, int TNt>
__global__ __launch_bounds__(BM * 2)
void conv_wmma_kernel(const float* __restrict__ X, const float* __restrict__ W,
                      const float* __restrict__ gamma, const float* __restrict__ beta,
                      const float* __restrict__ mean,  const float* __restrict__ var,
                      const float* __restrict__ residual, float* __restrict__ Out,
                      int Bn, int Cin, int Hin, int Win,
                      int Cout, int Hout, int Wout,
                      int kh, int kw, int stride, int pad, int relu)
{
    constexpr int NT = BM * 2;               // threads per block
    constexpr int NI = TNt / 16;             // 16x16 col-subtiles per wave
    constexpr int EB = (TNt * TK) / NT;      // B elems staged per thread

    __shared__ __align__(16) _Float16 sA[2][BM * LDA];
    __shared__ __align__(16) _Float16 sB[2][TNt * LDA];

    const int tid  = threadIdx.x;
    const int lane = tid & 31;
    const int wid  = tid >> 5;

    const int M      = Bn * Hout * Wout;
    const int K      = Cin * kh * kw;
    const int khw    = kh * kw;
    const int HinWin = Hin * Win;
    const int tileM  = blockIdx.x * BM;
    const int tileN  = blockIdx.y * TNt;

    // ---- per-thread A-stage mapping (fixed across K chunks)
    const int a_r    = tid % BM;            // tile row
    const int a_koff = (tid / BM) * 16;     // k offset within chunk: 0 or 16
    const int a_m    = tileM + a_r;
    const bool a_ok  = a_m < M;
    int hbase = 0, wbase = 0, xbase = 0;
    if (a_ok) {
        int wo = a_m % Wout;
        int t1 = a_m / Wout;
        int ho = t1 % Hout;
        int b  = t1 / Hout;
        hbase = ho * stride - pad;
        wbase = wo * stride - pad;
        xbase = b * Cin * HinWin;
    }

    const int nchunks = (K + TK - 1) / TK;

    _Float16 va[16];
    _Float16 vb[EB];

    // ---- global gather of one chunk into registers (fp32 -> f16)
    auto load_regs = [&](int k0) {
        int kg = k0 + a_koff;
        int ci = kg / khw;
        int rr = kg - ci * khw;
        int r  = rr / kw;
        int s  = rr - r * kw;
#pragma unroll
        for (int i = 0; i < 16; ++i) {
            float v = 0.0f;
            if (a_ok && (kg + i) < K) {
                int hi = hbase + r, wi = wbase + s;
                if (hi >= 0 && hi < Hin && wi >= 0 && wi < Win)
                    v = X[xbase + ci * HinWin + hi * Win + wi];
            }
            va[i] = (_Float16)v;
            if (++s == kw) { s = 0; if (++r == kh) { r = 0; ++ci; } }
        }
#pragma unroll
        for (int i = 0; i < EB; ++i) {
            int idx = tid * EB + i;
            int n   = idx >> 5;          // TK==32
            int kk  = idx & 31;
            int kgb = k0 + kk;
            float v = (kgb < K) ? W[(size_t)(tileN + n) * K + kgb] : 0.0f;
            vb[i] = (_Float16)v;
        }
    };
    auto store_regs = [&](int buf) {
        _Float16* pa = sA[buf] + a_r * LDA + a_koff;
#pragma unroll
        for (int i = 0; i < 16; ++i) pa[i] = va[i];
#pragma unroll
        for (int i = 0; i < EB; ++i) {
            int idx = tid * EB + i;
            sB[buf][(idx >> 5) * LDA + (idx & 31)] = vb[i];
        }
    };

    v8f acc[NI];
#pragma unroll
    for (int i = 0; i < NI; ++i)
#pragma unroll
        for (int j = 0; j < 8; ++j) acc[i][j] = 0.0f;

    load_regs(0);
    store_regs(0);

    const int frag_m  = wid * 16 + (lane & 15);
    const int frag_ka = (lane >> 4) * 8;
    const int frag_n0 = lane & 15;
    const int frag_kb = (lane >> 4) * 8;

    for (int ch = 0; ch < nchunks; ++ch) {
        __syncthreads();                       // tiles for chunk `ch` visible
        const int cur = ch & 1;
        if (ch + 1 < nchunks) load_regs((ch + 1) * TK);   // overlap with WMMAs

        Frag16 af;
        {
            const _Float16* p = sA[cur] + frag_m * LDA + frag_ka;
            af.q[0] = *(const f32x4*)(p);
            af.q[1] = *(const f32x4*)(p + 16);
        }
#pragma unroll
        for (int ni = 0; ni < NI; ++ni) {
            Frag16 bf;
            const _Float16* p = sB[cur] + (ni * 16 + frag_n0) * LDA + frag_kb;
            bf.q[0] = *(const f32x4*)(p);
            bf.q[1] = *(const f32x4*)(p + 16);
            acc[ni] = __builtin_amdgcn_wmma_f32_16x16x32_f16(
                false, af.v, false, bf.v, (short)0, acc[ni], false, false);
        }
        if (ch + 1 < nchunks) store_regs((ch + 1) & 1);   // other buffer: no race
    }

    // ---- epilogue: BN fold, residual, ReLU, scatter to NCHW
    const float eps = 1e-5f;
#pragma unroll
    for (int ni = 0; ni < NI; ++ni) {
        int n = tileN + ni * 16 + (lane & 15);
        float sc = gamma[n] / sqrtf(var[n] + eps);
        float sh = beta[n] - mean[n] * sc;
#pragma unroll
        for (int r = 0; r < 8; ++r) {
            int m = tileM + wid * 16 + r + ((lane >> 4) << 3);
            if (m >= M) continue;
            int wo = m % Wout;
            int t1 = m / Wout;
            int ho = t1 % Hout;
            int b  = t1 / Hout;
            float v = acc[ni][r] * sc + sh;
            size_t o = (((size_t)b * Cout + n) * Hout + ho) * Wout + wo;
            if (residual) v += residual[o];
            if (relu) v = v > 0.0f ? v : 0.0f;
            Out[o] = v;
        }
    }
}

// ---------------------------------------------------------------------------
// 3x3 stride-2 pad-1 max pool (-inf padding, matches reduce_window)
// ---------------------------------------------------------------------------
__global__ void maxpool_kernel(const float* __restrict__ in, float* __restrict__ out,
                               int Bn, int C, int Hin, int Win, int Hout, int Wout)
{
    int idx = blockIdx.x * blockDim.x + threadIdx.x;
    int total = Bn * C * Hout * Wout;
    if (idx >= total) return;
    int wo = idx % Wout; int t = idx / Wout;
    int ho = t % Hout;   t /= Hout;
    int c  = t % C;      int b = t / C;
    float m = -INFINITY;
    for (int r = 0; r < 3; ++r)
        for (int s = 0; s < 3; ++s) {
            int hi = ho * 2 - 1 + r, wi = wo * 2 - 1 + s;
            if (hi >= 0 && hi < Hin && wi >= 0 && wi < Win) {
                float v = in[((size_t)(b * C + c) * Hin + hi) * Win + wi];
                m = fmaxf(m, v);
            }
        }
    out[idx] = m;
}

// ---------------------------------------------------------------------------
// Global average pool: (B,C,H,W) -> (B,C)
// ---------------------------------------------------------------------------
__global__ void avgpool_kernel(const float* __restrict__ in, float* __restrict__ out,
                               int total, int HW)
{
    int idx = blockIdx.x * blockDim.x + threadIdx.x;
    if (idx >= total) return;
    const float* p = in + (size_t)idx * HW;
    float s = 0.0f;
    for (int i = 0; i < HW; ++i) s += p[i];
    out[idx] = s / (float)HW;
}

// ---------------------------------------------------------------------------
// angles = sigmoid(feats @ Wq^T + bq) * pi        (32x8)
// ---------------------------------------------------------------------------
__global__ void fc2q_kernel(const float* __restrict__ feats, const float* __restrict__ w,
                            const float* __restrict__ bias, float* __restrict__ angles)
{
    int idx = blockIdx.x * blockDim.x + threadIdx.x;
    if (idx >= 32 * 8) return;
    int q = idx & 7, b = idx >> 3;
    float s = bias[q];
    const float* wr = w + q * 512;
    const float* f  = feats + b * 512;
    for (int k = 0; k < 512; ++k) s += f[k] * wr[k];
    angles[idx] = 3.14159265358979323846f / (1.0f + expf(-s));
}

// ---------------------------------------------------------------------------
// 8-qubit statevector sim. wire 0 = MSB of flattened index (bit 7-q).
// One block per sample; state in LDS; RX = 128 disjoint pairs; CNOT = 64 swaps.
// ---------------------------------------------------------------------------
__device__ __forceinline__ void rx_pairs(float* re, float* im, int t, int q, float half_theta)
{
    float c = cosf(half_theta), s = sinf(half_theta);
    int pos = 7 - q;
    int i0 = ((t >> pos) << (pos + 1)) | (t & ((1 << pos) - 1));
    int i1 = i0 | (1 << pos);
    float ar = re[i0], ai = im[i0], br = re[i1], bi = im[i1];
    // [c, -i s; -i s, c]
    re[i0] = c * ar + s * bi;  im[i0] = c * ai - s * br;
    re[i1] = c * br + s * ai;  im[i1] = c * bi - s * ar;
}

__global__ __launch_bounds__(128)
void quantum_kernel(const float* __restrict__ angles, const float* __restrict__ qw,
                    float* __restrict__ qf)
{
    __shared__ float re[256], im[256];
    __shared__ float acc[8];
    int b = blockIdx.x;
    int t = threadIdx.x;

    for (int i = t; i < 256; i += 128) { re[i] = 0.0f; im[i] = 0.0f; }
    if (t == 0) re[0] = 1.0f;
    if (t < 8)  acc[t] = 0.0f;
    __syncthreads();

    // Angle embedding (per-sample RX)
    for (int q = 0; q < 8; ++q) {
        rx_pairs(re, im, t, q, angles[b * 8 + q] * 0.5f);
        __syncthreads();
    }
    // 5 entangler layers: RX(shared) on each wire, then ring CNOTs
    for (int l = 0; l < 5; ++l) {
        for (int q = 0; q < 8; ++q) {
            rx_pairs(re, im, t, q, qw[l * 8 + q] * 0.5f);
            __syncthreads();
        }
        for (int q = 0; q < 8; ++q) {
            int cp = 7 - q, tp = 7 - ((q + 1) & 7);
            if (t < 64) {
                int plo = cp < tp ? cp : tp;
                int phi = cp < tp ? tp : cp;
                int i = t;
                i = ((i >> plo) << (plo + 1)) | (i & ((1 << plo) - 1));
                i = ((i >> phi) << (phi + 1)) | (i & ((1 << phi) - 1));
                i |= (1 << cp);            // control=1, target=0
                int j = i | (1 << tp);
                float tr = re[i], ti = im[i];
                re[i] = re[j]; im[i] = im[j];
                re[j] = tr;    im[j] = ti;
            }
            __syncthreads();
        }
    }
    // <Z_q> = sum_k sign(k,q) * |amp_k|^2
    for (int i = t; i < 256; i += 128) {
        float p = re[i] * re[i] + im[i] * im[i];
        for (int q = 0; q < 8; ++q) {
            float sgn = ((i >> (7 - q)) & 1) ? -p : p;
            atomicAdd(&acc[q], sgn);
        }
    }
    __syncthreads();
    if (t < 8) qf[b * 8 + t] = acc[t];
}

// ---------------------------------------------------------------------------
// h = relu([feats, qf] @ W1^T + b1)   (32 x 256, K = 512 + 8)
// ---------------------------------------------------------------------------
__global__ void fc1_kernel(const float* __restrict__ feats, const float* __restrict__ qf,
                           const float* __restrict__ w, const float* __restrict__ bias,
                           float* __restrict__ h)
{
    int idx = blockIdx.x * blockDim.x + threadIdx.x;
    if (idx >= 32 * 256) return;
    int o = idx & 255, b = idx >> 8;
    const float* wr = w + o * 520;
    const float* f  = feats + b * 512;
    float s = bias[o];
    for (int k = 0; k < 512; ++k) s += f[k] * wr[k];
    for (int k = 0; k < 8;   ++k) s += qf[b * 8 + k] * wr[512 + k];
    h[idx] = s > 0.0f ? s : 0.0f;
}

// ---------------------------------------------------------------------------
// out = h @ W2^T + b2   (32 x 4)
// ---------------------------------------------------------------------------
__global__ void fc2_kernel(const float* __restrict__ h, const float* __restrict__ w,
                           const float* __restrict__ bias, float* __restrict__ out)
{
    int idx = blockIdx.x * blockDim.x + threadIdx.x;
    if (idx >= 32 * 4) return;
    int o = idx & 3, b = idx >> 2;
    float s = bias[o];
    const float* wr = w + o * 256;
    const float* hr = h + b * 256;
    for (int k = 0; k < 256; ++k) s += hr[k] * wr[k];
    out[idx] = s;
}

// ---------------------------------------------------------------------------
// Launcher. Input index map (setup_inputs dict insertion order; nested
// 'params' pytree flattened JAX-style: dict keys sorted alphabetically):
//   0:x  1-4:bn1{beta,gamma,mean,var}  5:conv1
//   stage0 b0: 6-9 bn1, 10-13 bn2, 14 conv1, 15 conv2
//   stage0 b1: 16-19, 20-23, 24, 25
//   stage1 b0: 26-29 bn1, 30-33 bn2, 34 conv1, 35 conv2, 36-39 down_bn, 40 down_conv
//   stage1 b1: 41-44, 45-48, 49, 50
//   stage2 b0: 51-54, 55-58, 59, 60, 61-64, 65
//   stage2 b1: 66-69, 70-73, 74, 75
//   stage3 b0: 76-79, 80-83, 84, 85, 86-89, 90
//   stage3 b1: 91-94, 95-98, 99, 100
//   101:fc2q_w 102:fc2q_b 103:q_weights 104:cls_w1 105:cls_b1 106:cls_w2 107:cls_b2
// ---------------------------------------------------------------------------
extern "C" void kernel_launch(void* const* d_in, const int* in_sizes, int n_in,
                              void* d_out, int out_size, void* d_ws, size_t ws_size,
                              hipStream_t stream)
{
    (void)in_sizes; (void)n_in; (void)out_size; (void)ws_size;
    auto IN = [&](int i) { return (const float*)d_in[i]; };

    // workspace carve-up (floats)
    float* ws = (float*)d_ws;
    size_t off = 0;
    float* buf0 = ws + off; off += (size_t)32 * 64 * 112 * 112;   // 25,690,112
    float* buf1 = ws + off; off += (size_t)32 * 64 * 56 * 56;     // 6,422,528
    float* buf2 = ws + off; off += (size_t)32 * 64 * 56 * 56;
    float* buf3 = ws + off; off += (size_t)32 * 64 * 56 * 56;
    float* buf4 = ws + off; off += (size_t)32 * 64 * 56 * 56;
    float* feats  = ws + off; off += 32 * 512;
    float* angles = ws + off; off += 32 * 8;
    float* qfbuf  = ws + off; off += 32 * 8;
    float* hbuf   = ws + off; off += 32 * 256;

    auto convbn = [&](const float* X, int widx, int bnbase, const float* res, float* out,
                      int Cin, int Hin, int Win, int Cout, int Hout, int Wout,
                      int k, int stride, int pad, int relu) {
        int M = 32 * Hout * Wout;
        if (M >= 4096 && Cout >= 128) {
            // wide-N tile: best A-reuse, 8 WMMAs per barrier per wave
            dim3 grid((M + 127) / 128, Cout / 128);
            conv_wmma_kernel<128, 128><<<grid, 256, 0, stream>>>(
                X, IN(widx), IN(bnbase + 1), IN(bnbase + 0), IN(bnbase + 2), IN(bnbase + 3),
                res, out, 32, Cin, Hin, Win, Cout, Hout, Wout, k, k, stride, pad, relu);
        } else if (M >= 4096) {
            dim3 grid((M + 127) / 128, Cout / 64);
            conv_wmma_kernel<128, 64><<<grid, 256, 0, stream>>>(
                X, IN(widx), IN(bnbase + 1), IN(bnbase + 0), IN(bnbase + 2), IN(bnbase + 3),
                res, out, 32, Cin, Hin, Win, Cout, Hout, Wout, k, k, stride, pad, relu);
        } else {
            dim3 grid((M + 63) / 64, Cout / 64);
            conv_wmma_kernel<64, 64><<<grid, 128, 0, stream>>>(
                X, IN(widx), IN(bnbase + 1), IN(bnbase + 0), IN(bnbase + 2), IN(bnbase + 3),
                res, out, 32, Cin, Hin, Win, Cout, Hout, Wout, k, k, stride, pad, relu);
        }
    };

    // stem
    convbn(IN(0), 5, 1, nullptr, buf0, 3, 224, 224, 64, 112, 112, 7, 2, 3, 1);
    {
        int total = 32 * 64 * 56 * 56;
        maxpool_kernel<<<(total + 255) / 256, 256, 0, stream>>>(buf0, buf1, 32, 64, 112, 112, 56, 56);
    }

    // stage 0 (64 -> 64, stride 1)
    convbn(buf1, 14, 6,  nullptr, buf2, 64, 56, 56, 64, 56, 56, 3, 1, 1, 1);
    convbn(buf2, 15, 10, buf1,    buf3, 64, 56, 56, 64, 56, 56, 3, 1, 1, 1);
    convbn(buf3, 24, 16, nullptr, buf2, 64, 56, 56, 64, 56, 56, 3, 1, 1, 1);
    convbn(buf2, 25, 20, buf3,    buf1, 64, 56, 56, 64, 56, 56, 3, 1, 1, 1);

    // stage 1 (64 -> 128, stride 2)
    convbn(buf1, 40, 36, nullptr, buf4, 64, 56, 56, 128, 28, 28, 1, 2, 0, 0);  // downsample
    convbn(buf1, 34, 26, nullptr, buf2, 64, 56, 56, 128, 28, 28, 3, 2, 1, 1);
    convbn(buf2, 35, 30, buf4,    buf3, 128, 28, 28, 128, 28, 28, 3, 1, 1, 1);
    convbn(buf3, 49, 41, nullptr, buf2, 128, 28, 28, 128, 28, 28, 3, 1, 1, 1);
    convbn(buf2, 50, 45, buf3,    buf1, 128, 28, 28, 128, 28, 28, 3, 1, 1, 1);

    // stage 2 (128 -> 256, stride 2)
    convbn(buf1, 65, 61, nullptr, buf4, 128, 28, 28, 256, 14, 14, 1, 2, 0, 0);
    convbn(buf1, 59, 51, nullptr, buf2, 128, 28, 28, 256, 14, 14, 3, 2, 1, 1);
    convbn(buf2, 60, 55, buf4,    buf3, 256, 14, 14, 256, 14, 14, 3, 1, 1, 1);
    convbn(buf3, 74, 66, nullptr, buf2, 256, 14, 14, 256, 14, 14, 3, 1, 1, 1);
    convbn(buf2, 75, 70, buf3,    buf1, 256, 14, 14, 256, 14, 14, 3, 1, 1, 1);

    // stage 3 (256 -> 512, stride 2)
    convbn(buf1, 90, 86, nullptr, buf4, 256, 14, 14, 512, 7, 7, 1, 2, 0, 0);
    convbn(buf1, 84, 76, nullptr, buf2, 256, 14, 14, 512, 7, 7, 3, 2, 1, 1);
    convbn(buf2, 85, 80, buf4,    buf3, 512, 7, 7, 512, 7, 7, 3, 1, 1, 1);
    convbn(buf3, 99, 91, nullptr, buf2, 512, 7, 7, 512, 7, 7, 3, 1, 1, 1);
    convbn(buf2, 100, 95, buf3,   buf1, 512, 7, 7, 512, 7, 7, 3, 1, 1, 1);

    // head
    avgpool_kernel<<<(32 * 512 + 255) / 256, 256, 0, stream>>>(buf1, feats, 32 * 512, 49);
    fc2q_kernel<<<1, 256, 0, stream>>>(feats, IN(101), IN(102), angles);
    quantum_kernel<<<32, 128, 0, stream>>>(angles, IN(103), qfbuf);
    fc1_kernel<<<(32 * 256 + 255) / 256, 256, 0, stream>>>(feats, qfbuf, IN(104), IN(105), hbuf);
    fc2_kernel<<<1, 128, 0, stream>>>(hbuf, IN(106), IN(107), (float*)d_out);
}
